// MyValModel_25890062860854
// MI455X (gfx1250) — matrix operand order repaired and checked by reading, not compiled, not run on hardware
//
#include <hip/hip_runtime.h>
#include <hip/hip_bf16.h>

// ---------------------------------------------------------------------------
// CDNA5 (gfx1250) WMMA / vector types
// ---------------------------------------------------------------------------
typedef __attribute__((ext_vector_type(16))) _Float16 v16h;
typedef __attribute__((ext_vector_type(8)))  float    v8f;
typedef __attribute__((ext_vector_type(4)))  float    f32x4;
typedef __attribute__((ext_vector_type(8)))  _Float16 f16x8;
typedef __attribute__((ext_vector_type(4)))  _Float16 f16x4;
// Matches the async-to-LDS builtin's parameter type exactly:
typedef int b128_t __attribute__((vector_size(16)));

#define GLOBAL_AS __attribute__((address_space(1)))
#define SHARED_AS __attribute__((address_space(3)))

#define WAVES_PER_BLOCK 8
#define BLOCK_ROWS      (WAVES_PER_BLOCK * 16)   // 128 output rows per block

// Async global->LDS path (CDNA5). Guarded so compilation never regresses.
#if defined(__AMDGCN__) &&                                                     \
    __has_builtin(__builtin_amdgcn_global_load_async_to_lds_b128) &&           \
    __has_builtin(__builtin_amdgcn_s_wait_asynccnt)
#define USE_ASYNC_LDS 1
#else
#define USE_ASYNC_LDS 0
#endif

// ---------------------------------------------------------------------------
// Load one 16x32 f16 A-fragment (rows = M, cols = K) from a row-major fp32
// matrix, in contiguous k order (same order as the B fragments):
//   a[j]   -> K = kBase + (lane/16)*16 + j        (j = 0..7)
//   a[8+j] -> K = kBase + (lane/16)*16 + 8 + j    (j = 0..7)
// Each run is 8 consecutive fp32 -> two aligned f32x4 loads (NT for adj).
// ---------------------------------------------------------------------------
template <bool NT>
__device__ inline v16h load_a_frag(const float* __restrict__ A, int lda,
                                   int rowBase, int kBase, int kLimit,
                                   int rowLimit) {
  const int lane = threadIdx.x & 31;
  int m = rowBase + (lane & 15);
  if (m >= rowLimit) m = rowLimit - 1;        // clamp; extra rows never stored
  const int hi = lane >> 4;
  const float* rp = A + (size_t)m * lda;
  const int k0 = kBase + hi * 16;
  const int k1 = k0 + 8;
  v16h a;
  if (k1 + 7 < kLimit) {                       // fast vector path (aligned)
    f32x4 f0, f1, f2, f3;
    if (NT) {
      f0 = __builtin_nontemporal_load((const f32x4*)(rp + k0));
      f1 = __builtin_nontemporal_load((const f32x4*)(rp + k0 + 4));
      f2 = __builtin_nontemporal_load((const f32x4*)(rp + k1));
      f3 = __builtin_nontemporal_load((const f32x4*)(rp + k1 + 4));
    } else {
      f0 = *(const f32x4*)(rp + k0);
      f1 = *(const f32x4*)(rp + k0 + 4);
      f2 = *(const f32x4*)(rp + k1);
      f3 = *(const f32x4*)(rp + k1 + 4);
    }
#pragma unroll
    for (int j = 0; j < 4; ++j) {
      a[j]      = (_Float16)f0[j];
      a[4 + j]  = (_Float16)f1[j];
      a[8 + j]  = (_Float16)f2[j];
      a[12 + j] = (_Float16)f3[j];
    }
  } else {                                     // K tail: element-wise guarded
#pragma unroll
    for (int j = 0; j < 8; ++j) {
      const int ka = k0 + j, kb = k1 + j;
      a[j]     = (_Float16)((ka < kLimit) ? rp[ka] : 0.0f);
      a[8 + j] = (_Float16)((kb < kLimit) ? rp[kb] : 0.0f);
    }
  }
  return a;
}

// ---------------------------------------------------------------------------
// Stage a 32 x (16*NCT) fp32 B-chunk into LDS as f16, pre-swizzled into WMMA
// per-lane order (used only by the small feature GEMMs for the weights):
//   element j of lane L, col-tile c  ->  B[(L/16)*16 + j][c*16 + L%16]
// ---------------------------------------------------------------------------
__device__ inline void stage_b(const float* __restrict__ Bm, int ldb,
                               int kBase, int kLimit, _Float16* sB, int ncolt) {
  const int ncols = ncolt * 16;
  const int total = 32 * ncols;
  for (int idx = threadIdx.x; idx < total; idx += blockDim.x) {
    const int k = idx / ncols;
    const int n = idx - k * ncols;
    const float v =
        (kBase + k < kLimit) ? Bm[(size_t)(kBase + k) * ldb + n] : 0.0f;
    const int c    = n >> 4;
    const int lane = ((k >> 4) << 4) | (n & 15);
    const int j    = k & 15;
    sB[((c * 32 + lane) << 4) + j] = (_Float16)v;
  }
}

__device__ inline v16h load_b_frag(const _Float16* sB, int c) {
  const int lane = threadIdx.x & 31;
  return *(const v16h*)(sB + ((c * 32 + lane) << 4));
}

// ---------------------------------------------------------------------------
// One-time pre-pass: convert+swizzle the dense-node matrix Bm [B,N,F] (fp32,
// row-major) into chunk-contiguous f16 WMMA fragments:
//   Bh[(b*nK + kc)*NCT*512 + c*512 + lane*16 + j]
//     = (f16) Bm[b][kc*32 + ((lane/16)*16 | j)][c*16 + lane%16]
// After this, hot-loop staging is a pure 128-bit copy.
// ---------------------------------------------------------------------------
template <int NCT>
__global__ void __launch_bounds__(256)
swizzle_b_kernel(const float* __restrict__ Bm, int N, int nK,
                 _Float16* __restrict__ Bh) {
  const int F = NCT * 16;
  const int kc = blockIdx.x;
  const int b  = blockIdx.y;
  const float* Bb = Bm + (size_t)b * N * F;
  _Float16* dst = Bh + ((size_t)b * nK + kc) * (NCT * 512);
  const int kBase = kc * 32;
  for (int o4 = threadIdx.x; o4 < NCT * 128; o4 += blockDim.x) {
    const int o    = o4 * 4;            // 4 consecutive halves (same lane)
    const int c    = o >> 9;
    const int lane = (o >> 4) & 31;
    const int j0   = o & 15;
    const int n    = c * 16 + (lane & 15);
    f16x4 t;
#pragma unroll
    for (int u = 0; u < 4; ++u) {
      const int k = kBase + (((lane >> 4) << 4) | (j0 + u));
      t[u] = (_Float16)((k < N) ? Bb[(size_t)k * F + n] : 0.0f);
    }
    *(f16x4*)(dst + o) = t;
  }
}

// ---------------------------------------------------------------------------
// Chunk staging: pre-swizzled f16 chunk (CHUNK halves) -> LDS.
// Async path: one GLOBAL_LOAD_ASYNC_TO_LDS_B128 per lane (ASYNCcnt).
// ---------------------------------------------------------------------------
template <int CHUNK>
__device__ inline void stage_chunk_async(const _Float16* __restrict__ src,
                                         _Float16* dst) {
#if USE_ASYNC_LDS
  for (int u = threadIdx.x; u < CHUNK / 8; u += blockDim.x) {
    __builtin_amdgcn_global_load_async_to_lds_b128(
        (GLOBAL_AS b128_t*)(src + u * 8),
        (SHARED_AS b128_t*)(dst + u * 8), 0, 0);
  }
#else
  (void)src; (void)dst;
#endif
}

template <int CHUNK>
__device__ inline void stage_chunk_sync(const _Float16* __restrict__ src,
                                        _Float16* dst) {
  for (int u = threadIdx.x; u < CHUNK / 8; u += blockDim.x) {
    *(f16x8*)(dst + u * 8) = *(const f16x8*)(src + u * 8);
  }
}

// ---------------------------------------------------------------------------
// Big kernel: C[b] = act( adj[b] @ Bm[b] + bias (+ skip[b]) )
// adj: [B,N,N] fp32 streamed NT; Bm comes in as pre-swizzled f16 chunks (Bh).
// 8 waves/block, each wave: 16 rows x 16*NCT cols, K-chunks of 32.
// Double-buffered LDS chunk, async copy overlapped one K-step ahead.
// ---------------------------------------------------------------------------
template <int NCT, bool RELU, bool HAS_SKIP>
__global__ void __launch_bounds__(256)
adj_spmm_kernel(const float* __restrict__ adj, int N, int nK,
                const _Float16* __restrict__ Bh,
                const float* __restrict__ bias,
                const float* __restrict__ skip, float* __restrict__ out) {
  constexpr int CHUNK = NCT * 512;               // halves per K-chunk
  __shared__ __align__(32) _Float16 sB[2][CHUNK];
  const int F = NCT * 16;
  const int b = blockIdx.y;
  const float* A = adj + (size_t)b * N * N;
  const _Float16* Bb = Bh + (size_t)b * nK * CHUNK;
  const float* Sb = HAS_SKIP ? (skip + (size_t)b * N * F) : nullptr;
  float* Ob = out + (size_t)b * N * F;

  const int wave = threadIdx.x >> 5;
  const int lane = threadIdx.x & 31;
  const int rowBase = blockIdx.x * BLOCK_ROWS + wave * 16;

  v8f acc[NCT];
#pragma unroll
  for (int c = 0; c < NCT; ++c) acc[c] = (v8f){0, 0, 0, 0, 0, 0, 0, 0};

#if USE_ASYNC_LDS
  stage_chunk_async<CHUNK>(Bb, sB[0]);           // prime the pipeline
  for (int kc = 0; kc < nK; ++kc) {
    const _Float16* cur = sB[kc & 1];
    __builtin_amdgcn_s_wait_asynccnt(0);         // my stage kc landed in LDS
    __syncthreads();                             // everyone's stage kc visible;
                                                 // also orders prior reads of
                                                 // the other buffer vs rewrite
    if (kc + 1 < nK)
      stage_chunk_async<CHUNK>(Bb + (size_t)(kc + 1) * CHUNK, sB[(kc + 1) & 1]);
    const v16h a = load_a_frag<true>(A, N, rowBase, kc * 32, N, N);
#pragma unroll
    for (int c = 0; c < NCT; ++c) {
      const v16h bf = load_b_frag(cur, c);
      acc[c] = __builtin_amdgcn_wmma_f32_16x16x32_f16(
          false, a, false, bf, (short)0, acc[c], false, false);
    }
  }
#else
  for (int kc = 0; kc < nK; ++kc) {
    __syncthreads();
    stage_chunk_sync<CHUNK>(Bb + (size_t)kc * CHUNK, sB[0]);
    __syncthreads();
    const v16h a = load_a_frag<true>(A, N, rowBase, kc * 32, N, N);
#pragma unroll
    for (int c = 0; c < NCT; ++c) {
      const v16h bf = load_b_frag(sB[0], c);
      acc[c] = __builtin_amdgcn_wmma_f32_16x16x32_f16(
          false, a, false, bf, (short)0, acc[c], false, false);
    }
  }
#endif

  // Epilogue: D layout -> lane holds col = lane%16, rows (lane/16)*8 + r
  const int n0   = lane & 15;
  const int mrow = rowBase + ((lane >> 4) << 3);
#pragma unroll
  for (int c = 0; c < NCT; ++c) {
    const int col = c * 16 + n0;
    const float bv = bias[col];
#pragma unroll
    for (int r = 0; r < 8; ++r) {
      const int row = mrow + r;
      if (row < N) {
        float v = acc[c][r] + bv;
        if (HAS_SKIP) v += Sb[(size_t)row * F + col];
        if (RELU) v = fmaxf(v, 0.0f);
        Ob[(size_t)row * F + col] = v;
      }
    }
  }
}

// ---------------------------------------------------------------------------
// Small feature GEMM: C[b] = X[b] @ W (+bias), X: [B,N,64], W: [64,16*NCT]
// Whole W staged (both K-chunks) in LDS once.
// ---------------------------------------------------------------------------
template <int NCT, bool HAS_BIAS>
__global__ void __launch_bounds__(256)
node_gemm_kernel(const float* __restrict__ X, int N,
                 const float* __restrict__ W, const float* __restrict__ bias,
                 float* __restrict__ out) {
  __shared__ __align__(32) _Float16 sW[2 * NCT * 32 * 16];
  const int F = NCT * 16;
  const int b = blockIdx.y;
  const float* Xb = X + (size_t)b * N * 64;
  float* Ob = out + (size_t)b * N * F;

  stage_b(W, F, 0, 64, sW, NCT);
  stage_b(W, F, 32, 64, sW + NCT * 512, NCT);
  __syncthreads();

  const int wave = threadIdx.x >> 5;
  const int lane = threadIdx.x & 31;
  const int rowBase = blockIdx.x * BLOCK_ROWS + wave * 16;

  v8f acc[NCT];
#pragma unroll
  for (int c = 0; c < NCT; ++c) acc[c] = (v8f){0, 0, 0, 0, 0, 0, 0, 0};

#pragma unroll
  for (int kc = 0; kc < 2; ++kc) {
    const v16h a = load_a_frag<false>(Xb, 64, rowBase, kc * 32, 64, N);
#pragma unroll
    for (int c = 0; c < NCT; ++c) {
      const v16h bf = load_b_frag(sW + kc * NCT * 512, c);
      acc[c] = __builtin_amdgcn_wmma_f32_16x16x32_f16(
          false, a, false, bf, (short)0, acc[c], false, false);
    }
  }

  const int n0   = lane & 15;
  const int mrow = rowBase + ((lane >> 4) << 3);
#pragma unroll
  for (int c = 0; c < NCT; ++c) {
    const int col = c * 16 + n0;
    const float bv = HAS_BIAS ? bias[col] : 0.0f;
#pragma unroll
    for (int r = 0; r < 8; ++r) {
      const int row = mrow + r;
      if (row < N) Ob[(size_t)row * F + col] = acc[c][r] + bv;
    }
  }
}

// ---------------------------------------------------------------------------
// Row max-pool: dst[b*64 + off + col] = max over rows of OUT[b][:, col]
// ---------------------------------------------------------------------------
__global__ void pool_kernel(const float* __restrict__ OUT, int N,
                            float* __restrict__ dst, int dstOff) {
  __shared__ float red[8][32];
  const int b = blockIdx.x;
  const float* P = OUT + (size_t)b * N * 32;
  const int col = threadIdx.x & 31;
  const int seg = threadIdx.x >> 5;
  float m = -3.402823466e38f;
  for (int r = seg; r < N; r += 8) m = fmaxf(m, P[(size_t)r * 32 + col]);
  red[seg][col] = m;
  __syncthreads();
  if (threadIdx.x < 32) {
    float mm = red[0][col];
#pragma unroll
    for (int s = 1; s < 8; ++s) mm = fmaxf(mm, red[s][col]);
    dst[b * 64 + dstOff + col] = mm;
  }
}

// ---------------------------------------------------------------------------
// Tiny MLP head on [2,64] -> [2,1]. One block, 64 threads (2 waves).
// ---------------------------------------------------------------------------
__global__ void mlp_head_kernel(const float* __restrict__ pooled,
                                const float* __restrict__ Wfc2, const float* __restrict__ bfc2,
                                const float* __restrict__ Wfc3, const float* __restrict__ bfc3,
                                const float* __restrict__ Wfc4, const float* __restrict__ bfc4,
                                const float* __restrict__ Wfc5, const float* __restrict__ bfc5,
                                float* __restrict__ out) {
  __shared__ float d0[2][64], h2[2][32], h3[2][64], h4[2][32];
  const int t = threadIdx.x;  // 0..63
  d0[0][t] = pooled[t];
  d0[1][t] = pooled[64 + t];
  __syncthreads();
  {  // fc2: [2,64] -> relu [2,32]
    const int b = t >> 5, j = t & 31;
    float s = bfc2[j];
    for (int k = 0; k < 64; ++k) s += d0[b][k] * Wfc2[k * 32 + j];
    h2[b][j] = fmaxf(s, 0.0f);
  }
  __syncthreads();
  for (int b = 0; b < 2; ++b) {  // fc3: [2,32] -> relu [2,64]
    float s = bfc3[t];
    for (int k = 0; k < 32; ++k) s += h2[b][k] * Wfc3[k * 64 + t];
    h3[b][t] = fmaxf(s, 0.0f);
  }
  __syncthreads();
  {  // fc4: [2,64] -> relu [2,32]
    const int b = t >> 5, j = t & 31;
    float s = bfc4[j];
    for (int k = 0; k < 64; ++k) s += h3[b][k] * Wfc4[k * 32 + j];
    h4[b][j] = fmaxf(s, 0.0f);
  }
  __syncthreads();
  if (t < 2) {  // fc5: [2,32] -> [2,1]
    float s = bfc5[0];
    for (int k = 0; k < 32; ++k) s += h4[t][k] * Wfc5[k];
    out[t] = s;
  }
}

// ---------------------------------------------------------------------------
// Launch
// ---------------------------------------------------------------------------
extern "C" void kernel_launch(void* const* d_in, const int* in_sizes, int n_in,
                              void* d_out, int out_size, void* d_ws,
                              size_t ws_size, hipStream_t stream) {
  (void)in_sizes; (void)n_in; (void)out_size; (void)ws_size;
  const float* solute_adj   = (const float*)d_in[0];
  const float* solute_meth  = (const float*)d_in[1];
  const float* solvent_meth = (const float*)d_in[2];
  const float* solvent_adj  = (const float*)d_in[3];
  const float* W_fc1 = (const float*)d_in[4];
  const float* b_fc1 = (const float*)d_in[5];
  const float* W_c1  = (const float*)d_in[6];
  const float* b_c1  = (const float*)d_in[7];
  const float* W_c2  = (const float*)d_in[8];
  const float* b_c2  = (const float*)d_in[9];
  const float* W_fc2 = (const float*)d_in[10];
  const float* b_fc2 = (const float*)d_in[11];
  const float* W_fc3 = (const float*)d_in[12];
  const float* b_fc3 = (const float*)d_in[13];
  const float* W_fc4 = (const float*)d_in[14];
  const float* b_fc4 = (const float*)d_in[15];
  const float* W_fc5 = (const float*)d_in[16];
  const float* b_fc5 = (const float*)d_in[17];
  float* out = (float*)d_out;

  const int B = 2;
  const int NMAX  = 8940;                  // buffers sized for larger branch
  const int nKmax = (NMAX + 31) / 32;      // 280

  char* w = (char*)d_ws;
  auto alloc = [&](size_t bytes) -> char* {
    char* p = w;
    w += (bytes + 255) & ~(size_t)255;
    return p;
  };
  float* P      = (float*)alloc((size_t)B * NMAX * 64 * 4);
  float* H      = (float*)alloc((size_t)B * NMAX * 64 * 4);
  float* Q      = (float*)alloc((size_t)B * NMAX * 32 * 4);
  float* INIT   = (float*)alloc((size_t)B * NMAX * 32 * 4);
  float* OUTb   = (float*)alloc((size_t)B * NMAX * 32 * 4);
  float* pooled = (float*)alloc(B * 64 * 4);
  _Float16* Bh64 = (_Float16*)alloc((size_t)B * nKmax * 2048 * 2);  // F=64 chunks
  _Float16* Bh32 = (_Float16*)alloc((size_t)B * nKmax * 1024 * 2);  // F=32 chunks

  struct Branch { const float* x; const float* adj; int N; int poolOff; };
  const Branch brs[2] = {
      {solute_meth, solute_adj, 2076, 0},
      {solvent_meth, solvent_adj, 8940, 32},
  };

  const dim3 blk(256);
  for (int i = 0; i < 2; ++i) {
    const int N  = brs[i].N;
    const int nK = (N + 31) / 32;
    const dim3 grid((N + BLOCK_ROWS - 1) / BLOCK_ROWS, B);
    const dim3 gridK(nK, B);
    // P = x @ W_c1                        [B,N,64]
    node_gemm_kernel<4, false><<<grid, blk, 0, stream>>>(brs[i].x, N, W_c1,
                                                         nullptr, P);
    // INIT = x @ W_fc1 + b_fc1            [B,N,32]
    node_gemm_kernel<2, true><<<grid, blk, 0, stream>>>(brs[i].x, N, W_fc1,
                                                        b_fc1, INIT);
    // Pre-swizzle P into f16 WMMA chunks
    swizzle_b_kernel<4><<<gridK, blk, 0, stream>>>(P, N, nK, Bh64);
    // H = relu(adj @ P + b_c1)            [B,N,64]  (adj pass 1, NT stream)
    adj_spmm_kernel<4, true, false><<<grid, blk, 0, stream>>>(
        brs[i].adj, N, nK, Bh64, b_c1, nullptr, H);
    // Q = H @ W_c2                        [B,N,32]
    node_gemm_kernel<2, false><<<grid, blk, 0, stream>>>(H, N, W_c2, nullptr,
                                                         Q);
    // Pre-swizzle Q into f16 WMMA chunks
    swizzle_b_kernel<2><<<gridK, blk, 0, stream>>>(Q, N, nK, Bh32);
    // OUT = adj @ Q + b_c2 + INIT         [B,N,32]  (adj pass 2, NT stream)
    adj_spmm_kernel<2, false, true><<<grid, blk, 0, stream>>>(
        brs[i].adj, N, nK, Bh32, b_c2, INIT, OUTb);
    // pooled[b, poolOff:poolOff+32] = max over rows
    pool_kernel<<<dim3(B), dim3(256), 0, stream>>>(OUTb, N, pooled,
                                                   brs[i].poolOff);
  }

  mlp_head_kernel<<<dim3(1), dim3(64), 0, stream>>>(
      pooled, W_fc2, b_fc2, W_fc3, b_fc3, W_fc4, b_fc4, W_fc5, b_fc5, out);
}
